// NeuralODERacingNetwork_47382079209556
// MI455X (gfx1250) — matrix operand order repaired and checked by reading, not compiled, not run on hardware
//
#include <hip/hip_runtime.h>
#include <stdint.h>

// ---------------------------------------------------------------------------
// Neural-ODE (RK4, 40 steps) persistent-LDS kernel for gfx1250 (MI455X).
// State y (f32) + RK accumulator (f32) stay resident in LDS for all 40 steps;
// GEMMs run on v_wmma_f32_16x16x32_bf16 with dual N-tile software pipelining;
// weights ping-pong through LDS half-matrix buffers, warmed with
// global_prefetch before the synchronous copy.
// ---------------------------------------------------------------------------

typedef __attribute__((ext_vector_type(16))) __bf16          v16bf;
typedef __attribute__((ext_vector_type(8)))  float           v8f;
typedef __attribute__((ext_vector_type(8)))  unsigned short  ush8;
typedef __attribute__((ext_vector_type(16))) unsigned short  ush16;

// ---- LDS layout (bytes) ----------------------------------------------------
#define OFF_CB1   0        // b1[256] f32
#define OFF_CS1   1024     // bn1 scale
#define OFF_CT1   2048     // bn1 shift
#define OFF_CB2   3072
#define OFF_CS2   4096
#define OFF_CT2   5120
#define OFF_CBO   6144
#define OFF_Y     7168     // 64 rows * 260 f32 (padded)   = 66560 B
#define OFF_ACC   73728    // 64 rows * 260 f32            = 66560 B
#define OFF_ACT   140288   // 64 rows * 264 bf16 (padded)  = 33792 B
#define OFF_WBUF  174080   // 2 half-matrix buffers, 128 rows * 264 bf16 each
#define WHALF_B   67584
#define LDS_TOTAL 309248   // <= 320KB WGP LDS

#define Y_STRIDE   260     // f32 per padded row  (1040 B, float4-aligned, bank-spread)
#define ACT_ROW_B  528     // bytes per padded bf16 row (264 halves)
#define W_ROW_B    528

__device__ __forceinline__ unsigned short f2bf(float f) {
  unsigned u = __builtin_bit_cast(unsigned, f);
  u += 0x7FFFu + ((u >> 16) & 1u);           // round-to-nearest-even
  return (unsigned short)(u >> 16);
}

__device__ __forceinline__ v16bf ld_frag(const char* p0, const char* p1) {
  ush8 a = *(const ush8*)p0;
  ush8 b = *(const ush8*)p1;
  ush16 u = __builtin_shufflevector(a, b, 0,1,2,3,4,5,6,7,8,9,10,11,12,13,14,15);
  return __builtin_bit_cast(v16bf, u);
}

__device__ __forceinline__ v8f wmma_bf16(v16bf a, v16bf b, v8f c) {
  // D = A(16x32) * B(32x16) + C, f32 accumulate
  return __builtin_amdgcn_wmma_f32_16x16x32_bf16(false, a, false, b, (short)0, c,
                                                 false, false);
}

__device__ __forceinline__ const char* phase_src(const unsigned short* wt, int ph) {
  const int seq = ph % 6;        // W1h0,W1h1,W2h0,W2h1,Woh0,Woh1
  const int mat = seq >> 1;
  const int n0  = (seq & 1) * 128;
  return (const char*)(wt + mat * 65536 + n0 * 256);
}

// Warm the near caches for phase ph's 64KB source region (hits L2) while the
// current GEMM half is executing.  Emits global_prefetch_b8.
__device__ __forceinline__ void warm_weights(const unsigned short* wt, int ph) {
  const char* src = phase_src(wt, ph);
  const int t = threadIdx.x;
#pragma unroll
  for (int i = 0; i < 4; ++i)                 // 512 x 128B lines / 128 threads
    __builtin_prefetch(src + (t * 4 + i) * 128, 0, 2);
}

// Stage one 64KB weight half (128 transposed rows of 512B payload) into a
// padded LDS ping-pong buffer.  Synchronous b128 copy (source pre-warmed).
__device__ __forceinline__ void copy_weights(const unsigned short* wt, int ph,
                                             char* smem) {
  const char* src = phase_src(wt, ph);
  char* dst = smem + OFF_WBUF + (ph & 1) * WHALF_B;
  const int t = threadIdx.x;
#pragma unroll
  for (int i = 0; i < 32; ++i) {
    int c   = t + i * 128;          // 4096 16-byte chunks total
    int row = c >> 5;
    int off = (c & 31) << 4;
    *(ush8*)(dst + row * W_ROW_B + off) = *(const ush8*)(src + row * 512 + off);
  }
}

// Fused per-16x16-tile epilogue (bias+ReLU+BN for layers 0/1; bias+tanh and the
// RK4 state update for layer 2), operating in the documented C-matrix layout.
__device__ __forceinline__ void tile_epilogue(
    int layer, int s, int nt, v8f c, int w, int side, int nlo, float dt,
    const float* CB1, const float* CS1, const float* CT1,
    const float* CB2, const float* CS2, const float* CT2, const float* CBO,
    float* yS, float* accS, unsigned short* actH) {
  int col = nt * 16 + nlo;
  if (layer < 2) {
    float bb = (layer == 0 ? CB1 : CB2)[col];
    float sc = (layer == 0 ? CS1 : CS2)[col];
    float tt = (layer == 0 ? CT1 : CT2)[col];
#pragma unroll
    for (int r = 0; r < 8; ++r) {
      int m = w * 16 + r + 8 * side;
      float z = fmaxf(c[r] + bb, 0.0f);
      actH[m * 264 + col] = f2bf(sc * z + tt);   // h -> next layer's A
    }
  } else {
    float bb = CBO[col];
#pragma unroll
    for (int r = 0; r < 8; ++r) {
      int m   = w * 16 + r + 8 * side;
      int idx = m * Y_STRIDE + col;
      float z  = c[r] + bb;
      float e  = __expf(z + z);                  // tanh(z)
      float kv = 1.0f - 2.0f * __builtin_amdgcn_rcpf(e + 1.0f);
      float yv = yS[idx];
      if (s == 3) {
        float yn = yv + (dt * (1.0f / 6.0f)) * (accS[idx] + kv);
        yS[idx] = yn;
        actH[m * 264 + col] = f2bf(yn);          // stage-0 input, next step
      } else {
        if (s == 0) accS[idx] = kv;
        else        accS[idx] += 2.0f * kv;
        float cc = (s == 2) ? dt : 0.5f * dt;
        actH[m * 264 + col] = f2bf(yv + cc * kv);
      }
    }
  }
}

// ---------------------------------------------------------------------------
// Prep kernel 1: transpose W (KxN,row-major f32) -> Wt (NxK, bf16) x3
__global__ void prep_weights_kernel(const float* __restrict__ W1,
                                    const float* __restrict__ W2,
                                    const float* __restrict__ Wo,
                                    unsigned short* __restrict__ wt) {
  int idx = blockIdx.x * 256 + threadIdx.x;     // 3*65536 total
  int m = idx >> 16;
  int rem = idx & 65535;
  int r = rem >> 8;        // k
  int c = rem & 255;       // n
  const float* W = (m == 0) ? W1 : ((m == 1) ? W2 : Wo);
  wt[m * 65536 + c * 256 + r] = f2bf(W[r * 256 + c]);
}

// Prep kernel 2: fold BatchNorm into per-channel scale/shift
__global__ void prep_consts_kernel(const float* g1, const float* be1,
                                   const float* m1, const float* v1,
                                   const float* g2, const float* be2,
                                   const float* m2, const float* v2,
                                   float* __restrict__ out) {
  int t = threadIdx.x;  // 256
  float s1 = g1[t] * rsqrtf(v1[t] + 1e-3f);
  out[t]       = s1;
  out[256 + t] = be1[t] - m1[t] * s1;
  float s2 = g2[t] * rsqrtf(v2[t] + 1e-3f);
  out[512 + t] = s2;
  out[768 + t] = be2[t] - m2[t] * s2;
}

// ---------------------------------------------------------------------------
__global__ __launch_bounds__(128)
void node_rk4_kernel(const float* __restrict__ y0,
                     const unsigned short* __restrict__ wt,
                     const float* __restrict__ cpack,
                     const float* __restrict__ b1,
                     const float* __restrict__ b2,
                     const float* __restrict__ bo,
                     float* __restrict__ out) {
  __shared__ __align__(16) char smem[LDS_TOTAL];

  const int tid  = threadIdx.x;
  const int lane = tid & 31;
  const int w    = tid >> 5;                 // wave id 0..3, owns rows 16w..16w+15
  const int blk  = blockIdx.x;               // 1024 blocks * 64 rows

  float* CB1 = (float*)(smem + OFF_CB1);
  float* CS1 = (float*)(smem + OFF_CS1);
  float* CT1 = (float*)(smem + OFF_CT1);
  float* CB2 = (float*)(smem + OFF_CB2);
  float* CS2 = (float*)(smem + OFF_CS2);
  float* CT2 = (float*)(smem + OFF_CT2);
  float* CBO = (float*)(smem + OFF_CBO);
  float* yS   = (float*)(smem + OFF_Y);
  float* accS = (float*)(smem + OFF_ACC);
  char*  actB = smem + OFF_ACT;
  unsigned short* actH = (unsigned short*)actB;

  // --- init: constants + resident y tile (f32) + bf16 copy for GEMM A ------
  for (int t = tid; t < 256; t += 128) {
    CB1[t] = b1[t];  CB2[t] = b2[t];  CBO[t] = bo[t];
    CS1[t] = cpack[t];        CT1[t] = cpack[256 + t];
    CS2[t] = cpack[512 + t];  CT2[t] = cpack[768 + t];
  }
  const float4* y0v = (const float4*)(y0 + (size_t)blk * 16384);
  for (int i = tid; i < 4096; i += 128) {
    float4 v = y0v[i];
    int row = i >> 6, c4 = i & 63;
    *(float4*)(yS + row * Y_STRIDE + c4 * 4) = v;
    unsigned short* ap = actH + row * 264 + c4 * 4;
    ap[0] = f2bf(v.x); ap[1] = f2bf(v.y); ap[2] = f2bf(v.z); ap[3] = f2bf(v.w);
  }
  copy_weights(wt, 0, smem);
  copy_weights(wt, 1, smem);
  __syncthreads();

  const float dt = 1.0f / 40.0f;
  int ph = 2;          // next phase to copy in
  int cp = 0;          // phase currently consumed

  for (int step = 0; step < 40; ++step) {
    for (int s = 0; s < 4; ++s) {            // RK4 stages
      for (int layer = 0; layer < 3; ++layer) {
        // Preload all A fragments (wave's 16x256 strip) — reused for 16 N-tiles.
        v16bf Af[8];
        {
          int row  = w * 16 + (lane & 15);
          int side = lane >> 4;
          const char* base = actB + row * ACT_ROW_B + side * 16;
#pragma unroll
          for (int kb = 0; kb < 8; ++kb)
            Af[kb] = ld_frag(base + kb * 64, base + kb * 64 + 32);
        }
        for (int half = 0; half < 2; ++half) {
          __syncthreads();                   // weight half ready / safe to read
          warm_weights(wt, ph);              // pull next half toward the WGP
          const char* wb = smem + OFF_WBUF + (cp & 1) * WHALF_B;
          const int side = lane >> 4;
          const int nlo  = lane & 15;
          // Two N-tiles per iteration: two independent accumulator chains and
          // two B-fragment streams keep 4 ds_load_b128 in flight per WMMA pair.
          for (int ntp = 0; ntp < 4; ++ntp) {
            v8f c0 = {0.f, 0.f, 0.f, 0.f, 0.f, 0.f, 0.f, 0.f};
            v8f c1 = {0.f, 0.f, 0.f, 0.f, 0.f, 0.f, 0.f, 0.f};
            const char* bb0 = wb + ((ntp * 2) * 16 + nlo) * W_ROW_B + side * 16;
            const char* bb1 = bb0 + 16 * W_ROW_B;
#pragma unroll
            for (int kb = 0; kb < 8; ++kb) {
              v16bf bf0 = ld_frag(bb0 + kb * 64, bb0 + kb * 64 + 32);
              v16bf bf1 = ld_frag(bb1 + kb * 64, bb1 + kb * 64 + 32);
              c0 = wmma_bf16(Af[kb], bf0, c0);
              c1 = wmma_bf16(Af[kb], bf1, c1);
            }
            int ntg = half * 8 + ntp * 2;
            tile_epilogue(layer, s, ntg,     c0, w, side, nlo, dt,
                          CB1, CS1, CT1, CB2, CS2, CT2, CBO, yS, accS, actH);
            tile_epilogue(layer, s, ntg + 1, c1, w, side, nlo, dt,
                          CB1, CS1, CT1, CB2, CS2, CT2, CBO, yS, accS, actH);
          }
          __syncthreads();                   // everyone done with this buffer
          copy_weights(wt, ph, smem);        // overwrite it with phase ph (warm)
          ++ph; ++cp;
        }
      }
    }
  }

  __syncthreads();
  float4* outv = (float4*)(out + (size_t)blk * 16384);
  for (int i = tid; i < 4096; i += 128) {
    int row = i >> 6, c4 = i & 63;
    outv[i] = *(const float4*)(yS + row * Y_STRIDE + c4 * 4);
  }
}

// ---------------------------------------------------------------------------
extern "C" void kernel_launch(void* const* d_in, const int* in_sizes, int n_in,
                              void* d_out, int out_size, void* d_ws, size_t ws_size,
                              hipStream_t stream) {
  const float* y0  = (const float*)d_in[0];
  const float* W1  = (const float*)d_in[1];
  const float* b1  = (const float*)d_in[2];
  const float* g1  = (const float*)d_in[3];
  const float* be1 = (const float*)d_in[4];
  const float* m1  = (const float*)d_in[5];
  const float* v1  = (const float*)d_in[6];
  const float* W2  = (const float*)d_in[7];
  const float* b2  = (const float*)d_in[8];
  const float* g2  = (const float*)d_in[9];
  const float* be2 = (const float*)d_in[10];
  const float* m2  = (const float*)d_in[11];
  const float* v2  = (const float*)d_in[12];
  const float* Wo  = (const float*)d_in[13];
  const float* bo  = (const float*)d_in[14];

  // workspace: 3 x (256x256 bf16 transposed weights) + 4x256 f32 BN consts
  unsigned short* wt = (unsigned short*)d_ws;
  float* cpack = (float*)((char*)d_ws + 3u * 65536u * 2u);

  prep_weights_kernel<<<768, 256, 0, stream>>>(W1, W2, Wo, wt);
  prep_consts_kernel<<<1, 256, 0, stream>>>(g1, be1, m1, v1, g2, be2, m2, v2, cpack);
  node_rk4_kernel<<<1024, 128, 0, stream>>>(y0, wt, cpack, b1, b2, bo, (float*)d_out);
}
